// Attention_34342558499032
// MI455X (gfx1250) — matrix operand order repaired and checked by reading, not compiled, not run on hardware
//
#include <hip/hip_runtime.h>

#define B_ 16
#define L_ 1024
#define D_ 256
#define EPSF 1e-7f
#define LOG2E 1.4426950408889634f

typedef __attribute__((ext_vector_type(16))) __bf16 v16bf;
typedef __attribute__((ext_vector_type(8)))  float  v8f;

__device__ __forceinline__ unsigned short f2bf(float f) {
  unsigned u = __float_as_uint(f);
  u += 0x7FFFu + ((u >> 16) & 1u);     // round-to-nearest-even
  return (unsigned short)(u >> 16);
}

// fast exp via v_exp_f32 (exp2)
__device__ __forceinline__ float fexp(float x) {
  return __builtin_amdgcn_exp2f(x * LOG2E);
}

// exp(tanh(e)) with bare TRANS ops: tanh(e) = 1 - 2/(1+exp(2e)); saturates
// correctly at +-1 because v_rcp_f32(inf)=0.
__device__ __forceinline__ float exp_tanh(float e) {
  float ex2 = __builtin_amdgcn_exp2f(e * (2.0f * LOG2E));
  float t   = 1.0f - 2.0f * __builtin_amdgcn_rcpf(1.0f + ex2);
  return __builtin_amdgcn_exp2f(t * LOG2E);
}

// A fragment, 16x32 bf16, chunk row stride 64 elements.
// lane L holds row M=L%16; lanes<16: K=[kb..kb+7] & [kb+16..kb+23] with kb=kk;
// lanes>=16: same with kb=kk+8.  (ISA 7.12.2, 16-bit A 16x32)
__device__ __forceinline__ v16bf load_a_frag(const unsigned short* X, int row, int kbase) {
  v16bf a;
  uint4* ap = (uint4*)&a;
  ap[0] = *(const uint4*)(X + row * 64 + kbase);
  ap[1] = *(const uint4*)(X + row * 64 + kbase + 16);
  return a;
}

// B fragment, 32x16 bf16 (K x N), B[k][n] = Xm[n][k].
// lane L holds column N=L%16; lanes<16: K=[kb..kb+7] & [kb+16..kb+23] with kb=kk; lanes>=16: kb=kk+8.
__device__ __forceinline__ v16bf load_b_frag(const unsigned short* X, int row, int kbase) {
  v16bf b;
  uint4* bp = (uint4*)&b;
  bp[0] = *(const uint4*)(X + row * 64 + kbase);
  bp[1] = *(const uint4*)(X + row * 64 + kbase + 16);
  return b;
}

__global__ void __launch_bounds__(256)
gram_attn_kernel(const float* __restrict__ x1, const float* __restrict__ x2,
                 float* __restrict__ colsum, float* __restrict__ wsum1,
                 float* __restrict__ wsum2, float* __restrict__ rowsum) {
  __shared__ unsigned short Xl1c[64 * 64], Xl2c[64 * 64], Xm1c[64 * 64], Xm2c[64 * 64];
  __shared__ float s1s[64], s2s[64], rowAcc[64], colAcc[64], w1Acc[64], w2Acc[64];

  const int b    = blockIdx.y;
  const int l0   = blockIdx.x * 64;
  const int tid  = threadIdx.x;
  const int w    = tid >> 5;
  const int lane = tid & 31;
  const int half = lane >> 4;
  const int ln15 = lane & 15;
  const int slb  = (w >> 2) * 32;   // waves share B-fragments across the i-subtiles
  const int sm   = (w & 3) * 16;

  const float* x1b = x1 + ((size_t)b * L_ + l0) * D_;
  const float* x2b = x2 + ((size_t)b * L_ + l0) * D_;

  if (tid < 64) { s1s[tid] = 0.f; s2s[tid] = 0.f; rowAcc[tid] = 0.f; }
  __syncthreads();

  { // per-row fp32 sums over D (4 threads per row, 64 elements each)
    int j = tid >> 2, seg = (tid & 3) << 6;
    const float* p1 = x1b + j * D_ + seg;
    const float* p2 = x2b + j * D_ + seg;
    float a1 = 0.f, a2 = 0.f;
    for (int q = 0; q < 64; ++q) { a1 += p1[q]; a2 += p2[q]; }
    atomicAdd(&s1s[j], a1);
    atomicAdd(&s2s[j], a2);
  }
  __syncthreads();

  // preload the s-weights this lane's accumulator rows need (C/D layout: VGPR r,
  // lanes 0-15 -> M=r, lanes 16-31 -> M=r+8)
  float s1v[2][8], s2v[2][8];
  for (int i = 0; i < 2; ++i)
    for (int r = 0; r < 8; ++r) {
      int lr = slb + i * 16 + r + half * 8;
      s1v[i][r] = s1s[lr];
      s2v[i][r] = s2s[lr];
    }

  float racc[2][8] = {};   // row-sum partials, persist over the whole m loop

  for (int m0 = 0; m0 < L_; m0 += 64) {
    if (tid < 64) { colAcc[tid] = 0.f; w1Acc[tid] = 0.f; w2Acc[tid] = 0.f; }
    const float* x1m = x1 + ((size_t)b * L_ + m0) * D_;
    const float* x2m = x2 + ((size_t)b * L_ + m0) * D_;

    v8f c12[2] = {}, c11[2] = {}, c22[2] = {};

    for (int k0 = 0; k0 < D_; k0 += 64) {
      __syncthreads();   // previous chunk consumed / accumulator resets visible
      for (int p = tid; p < 64 * 32; p += 256) {
        int row = p >> 5, kk = (p & 31) << 1;
        const float2 v1 = *(const float2*)(x1b + row * D_ + k0 + kk);
        const float2 v2 = *(const float2*)(x2b + row * D_ + k0 + kk);
        const float2 u1 = *(const float2*)(x1m + row * D_ + k0 + kk);
        const float2 u2 = *(const float2*)(x2m + row * D_ + k0 + kk);
        // prefetch next K-chunk while this one converts (global_prefetch_b8)
        if (k0 + 64 < D_) {
          __builtin_prefetch(x1m + row * D_ + k0 + 64 + kk, 0, 3);
          __builtin_prefetch(x2m + row * D_ + k0 + 64 + kk, 0, 3);
        }
        *(unsigned*)(Xl1c + row * 64 + kk) = (unsigned)f2bf(v1.x) | ((unsigned)f2bf(v1.y) << 16);
        *(unsigned*)(Xl2c + row * 64 + kk) = (unsigned)f2bf(v2.x) | ((unsigned)f2bf(v2.y) << 16);
        *(unsigned*)(Xm1c + row * 64 + kk) = (unsigned)f2bf(u1.x) | ((unsigned)f2bf(u1.y) << 16);
        *(unsigned*)(Xm2c + row * 64 + kk) = (unsigned)f2bf(u2.x) | ((unsigned)f2bf(u2.y) << 16);
      }
      __syncthreads();

      for (int kk = 0; kk < 64; kk += 32) {
        const int brow = sm + ln15;
        const int bkb  = kk + half * 8;
        v16bf bf1 = load_b_frag(Xm1c, brow, bkb);
        v16bf bf2 = load_b_frag(Xm2c, brow, bkb);
        for (int i = 0; i < 2; ++i) {
          const int arow = slb + i * 16 + ln15;
          const int akb  = kk + half * 8;
          v16bf af1 = load_a_frag(Xl1c, arow, akb);
          v16bf af2 = load_a_frag(Xl2c, arow, akb);
          c12[i] = __builtin_amdgcn_wmma_f32_16x16x32_bf16(false, af1, false, bf2, (short)0, c12[i], false, false);
          c11[i] = __builtin_amdgcn_wmma_f32_16x16x32_bf16(false, af1, false, bf1, (short)0, c11[i], false, false);
          c22[i] = __builtin_amdgcn_wmma_f32_16x16x32_bf16(false, af2, false, bf2, (short)0, c22[i], false, false);
        }
      }
    }

    // fused epilogue with bare TRANS intrinsics (v_sqrt/v_rcp/v_exp), which
    // co-execute with the XDL/WMMA pipe.
    for (int i = 0; i < 2; ++i) {
      float colp = 0.f, w1p = 0.f, w2p = 0.f;
      for (int r = 0; r < 8; ++r) {
        float den = __builtin_amdgcn_sqrtf(c11[i][r] * c22[i][r]) + EPSF;
        float e   = c12[i][r] * __builtin_amdgcn_rcpf(den);
        float a   = exp_tanh(e);
        racc[i][r] += a;
        colp += a;
        w1p += s1v[i][r] * a;
        w2p += s2v[i][r] * a;
      }
      int mc = sm + ln15;
      atomicAdd(&colAcc[mc], colp);
      atomicAdd(&w1Acc[mc], w1p);
      atomicAdd(&w2Acc[mc], w2p);
    }
    __syncthreads();
    if (tid < 64) {
      int gi = b * L_ + m0 + tid;
      atomicAdd(&colsum[gi], colAcc[tid]);
      atomicAdd(&wsum1[gi], w1Acc[tid]);
      atomicAdd(&wsum2[gi], w2Acc[tid]);
    }
  }

  // row sums: cross-lane reduce within 16-lane groups (xor 1,2,4,8 never crosses
  // the half boundary), then combine the 4 column-waves through LDS.
  for (int i = 0; i < 2; ++i)
    for (int r = 0; r < 8; ++r) {
      float v = racc[i][r];
      v += __shfl_xor(v, 1);
      v += __shfl_xor(v, 2);
      v += __shfl_xor(v, 4);
      v += __shfl_xor(v, 8);
      if (ln15 == 0) atomicAdd(&rowAcc[slb + i * 16 + r + half * 8], v);
    }
  __syncthreads();
  if (tid < 64) rowsum[b * L_ + l0 + tid] = rowAcc[tid];  // exclusive owner: plain store
}

__global__ void zero_ws(float* __restrict__ w, int n) {
  int i = blockIdx.x * blockDim.x + threadIdx.x;
  if (i < n) w[i] = 0.f;
}

__global__ void finalize_kernel(const float* __restrict__ colsum, const float* __restrict__ wsum1,
                                const float* __restrict__ wsum2, const float* __restrict__ rowsum,
                                float* __restrict__ out) {
  int i = blockIdx.x * blockDim.x + threadIdx.x;
  if (i < B_ * L_) {
    out[i]           = wsum2[i] / (colsum[i] + EPSF);   // ot1
    out[B_ * L_ + i] = wsum1[i] / (rowsum[i] + EPSF);   // ot2
  }
}

extern "C" void kernel_launch(void* const* d_in, const int* in_sizes, int n_in,
                              void* d_out, int out_size, void* d_ws, size_t ws_size,
                              hipStream_t stream) {
  const float* x1 = (const float*)d_in[0];
  const float* x2 = (const float*)d_in[1];
  float* W      = (float*)d_ws;
  float* colsum = W;
  float* wsum1  = W + (size_t)B_ * L_;
  float* wsum2  = W + (size_t)2 * B_ * L_;
  float* rowsum = W + (size_t)3 * B_ * L_;

  zero_ws<<<(3 * B_ * L_ + 255) / 256, 256, 0, stream>>>(W, 3 * B_ * L_);
  gram_attn_kernel<<<dim3(L_ / 64, B_), 256, 0, stream>>>(x1, x2, colsum, wsum1, wsum2, rowsum);
  finalize_kernel<<<(B_ * L_ + 255) / 256, 256, 0, stream>>>(colsum, wsum1, wsum2, rowsum, (float*)d_out);
}
